// TAttention_6700148982532
// MI455X (gfx1250) — matrix operand order repaired and checked by reading, not compiled, not run on hardware
//
#include <hip/hip_runtime.h>
#include <hip/hip_bf16.h>
#include <stdint.h>

#define B_  16
#define N_  2048
#define C_  512

typedef __attribute__((ext_vector_type(16))) __bf16 v16bf;
typedef __attribute__((ext_vector_type(8)))  __bf16 v8bf;
typedef __attribute__((ext_vector_type(8)))  float  v8f;

__device__ __forceinline__ unsigned short f2bf(float f) {
    union { float f; unsigned u; } v; v.f = f;
    unsigned u = v.u;
    return (unsigned short)((u + 0x7FFFu + ((u >> 16) & 1u)) >> 16);
}

// A-fragment (16x32 bf16): lanes 0-15 hold K{0..15},
// lanes 16-31 hold K{16..31}. khA = (lane>>4)*8.
__device__ __forceinline__ v16bf ldA(const unsigned short* rowk, int khA) {
    return *reinterpret_cast<const v16bf*>(rowk + 2 * khA);
}

// B-fragment (32x16 bf16): lane = column, 16 contiguous K values.
__device__ __forceinline__ v16bf ldB(const unsigned short* colk) {
    return *reinterpret_cast<const v16bf*>(colk);
}

__device__ __forceinline__ v8f vzero() {
    v8f z = {0.f, 0.f, 0.f, 0.f, 0.f, 0.f, 0.f, 0.f};
    return z;
}

#define WMMA_BF16(a, b, c) \
    __builtin_amdgcn_wmma_f32_16x16x32_bf16(false, (a), false, (b), (short)0, (c), false, false)

// gfx1250 async copy global -> LDS (ASYNCcnt-tracked, no VGPR data path).
// VDST = 32-bit LDS byte address, VADDR = 64-bit global address.
__device__ __forceinline__ void async_ld_b128(unsigned int lds_addr, const void* gaddr) {
    asm volatile("global_load_async_to_lds_b128 %0, %1, off"
                 :: "v"(lds_addr), "v"(gaddr) : "memory");
}
__device__ __forceinline__ void wait_async0() {
    asm volatile("s_wait_asynccnt 0" ::: "memory");
}

// ---------------------------------------------------------------------------
// K1: We[branch][t] = W3[:,:,t] @ W1  (per branch), cast to bf16. 9*C*C outs.
// ---------------------------------------------------------------------------
__global__ void __launch_bounds__(256)
prep_we(const float* __restrict__ w1q, const float* __restrict__ w3q,
        const float* __restrict__ w1k, const float* __restrict__ w3k,
        const float* __restrict__ w1v, const float* __restrict__ w3v,
        unsigned short* __restrict__ We) {
    int idx = blockIdx.x * 256 + threadIdx.x;          // 9*C*C threads
    int c = idx % C_;
    int o = (idx / C_) % C_;
    int w = idx / (C_ * C_);                           // 0..8
    int br = w / 3, t = w % 3;
    const float* w1 = (br == 0) ? w1q : (br == 1) ? w1k : w1v;
    const float* w3 = (br == 0) ? w3q : (br == 1) ? w3k : w3v;
    float s = 0.f;
    for (int m = 0; m < C_; ++m)
        s = fmaf(w3[(o * C_ + m) * 3 + t], w1[m * C_ + c], s);
    We[(size_t)w * C_ * C_ + (size_t)o * C_ + c] = f2bf(s);
}

// ---------------------------------------------------------------------------
// K2: f32 -> bf16, 4 elements per thread.
// ---------------------------------------------------------------------------
__global__ void __launch_bounds__(256)
cvt_bf4(const float* __restrict__ in, unsigned short* __restrict__ out, int n4) {
    int i = blockIdx.x * 256 + threadIdx.x;
    if (i >= n4) return;
    const float4 f = reinterpret_cast<const float4*>(in)[i];
    unsigned long long pk =  (unsigned long long)f2bf(f.x)
                          | ((unsigned long long)f2bf(f.y) << 16)
                          | ((unsigned long long)f2bf(f.z) << 32)
                          | ((unsigned long long)f2bf(f.w) << 48);
    reinterpret_cast<unsigned long long*>(out)[i] = pk;
}

// ---------------------------------------------------------------------------
// K3: fused (conv1x1 ∘ conv3 circular) branch as 3 shifted GEMMs:
//   out[b,o,n] = sum_t sum_c We[t][o,c] * xbf[b, (n+t-1) mod N, c]
// WG = 256 thr (8 waves): 2 M-subtiles x 4 N-subtiles, wave tile 16x64.
// Per tap t, the full 32x512 A-panel is staged into LDS with
// GLOBAL_LOAD_ASYNC_TO_LDS_B128 (one barrier pair per tap, not per K-step),
// then 16 K-steps run back-to-back with only B-operand global traffic.
// transposed=0 -> out[b][o][n] ([C][N]); transposed=1 -> out[b][n][o] ([N][C]).
// ---------------------------------------------------------------------------
__global__ void __launch_bounds__(256)
qkv_gemm(const unsigned short* __restrict__ xbf,   // [B][N][C] bf16
         const unsigned short* __restrict__ We,    // [3][C][C] bf16 (this branch)
         unsigned short* __restrict__ out,
         int transposed) {
    __shared__ __attribute__((aligned(32))) unsigned short lA[32][C_];  // 32 KiB
    const int tid  = threadIdx.x;
    const int lane = tid & 31;
    const int wave = tid >> 5;
    const int msub = wave >> 2;            // 0..1
    const int nsub = wave & 3;             // 0..3
    int bi = blockIdx.x;
    const int nb = bi % (N_ / 256); bi /= (N_ / 256);
    const int cb = bi % (C_ / 32);  bi /= (C_ / 32);
    const int b  = bi;
    const int Mbase = cb * 32;
    const int Nbase = nb * 256 + nsub * 64;
    const int l15 = lane & 15;
    const int khA = (lane >> 4) * 8;
    const int khB = (lane >> 4) * 16;
    const size_t xb = (size_t)b * N_ * C_;

    // async panel fill: thread -> (row fr, 128B starting at col fc)
    const int fr = tid >> 3;               // 0..31
    const int fc = (tid & 7) * 64;         // 0,64,...,448 (elements)

    v8f acc[4];
#pragma unroll
    for (int i = 0; i < 4; ++i) acc[i] = vzero();

    for (int t = 0; t < 3; ++t) {
        const unsigned short* Wt =
            We + (size_t)t * C_ * C_ + (size_t)(Mbase + fr) * C_ + fc;
        __syncthreads();                   // previous panel fully consumed
#pragma unroll
        for (int u = 0; u < 8; ++u) {
            async_ld_b128((unsigned int)(uintptr_t)&lA[fr][fc + u * 8],
                          Wt + u * 8);
        }
        wait_async0();                     // my async copies landed in LDS
        __syncthreads();                   // panel published to all waves

        // circular-shift row base pointers for this tap (hoisted out of K loop)
        const unsigned short* bp[4];
#pragma unroll
        for (int i = 0; i < 4; ++i) {
            int n  = Nbase + i * 16 + l15;
            int ns = n + t - 1;
            ns = (ns < 0) ? ns + N_ : (ns >= N_ ? ns - N_ : ns);
            bp[i] = xbf + xb + (size_t)ns * C_ + khB;
        }

        for (int kk = 0; kk < C_; kk += 32) {
            v16bf a = ldA(&lA[msub * 16 + l15][kk], khA);
#pragma unroll
            for (int i = 0; i < 4; ++i)
                acc[i] = WMMA_BF16(a, ldB(bp[i] + kk), acc[i]);
        }
    }

    const int Mtile  = Mbase + msub * 16;
    const int rowoff = (lane >> 4) * 8;    // D: lanes>=16 hold M = 8+j
#pragma unroll
    for (int i = 0; i < 4; ++i) {
        const int col = Nbase + i * 16 + l15;
#pragma unroll
        for (int j = 0; j < 8; ++j) {
            const int row = Mtile + rowoff + j;
            if (transposed)
                out[xb + (size_t)col * C_ + row] = f2bf(acc[i][j]);
            else
                out[(size_t)b * C_ * N_ + (size_t)row * N_ + col] = f2bf(acc[i][j]);
        }
    }
}

// ---------------------------------------------------------------------------
// K4: scores S[b,c,d] = (1/C) * sum_n q[b,c,n] * k[b,d,n]   (K = N = 2048)
// ---------------------------------------------------------------------------
__global__ void __launch_bounds__(256)
score_gemm(const unsigned short* __restrict__ q,   // [B][C][N] bf16
           const unsigned short* __restrict__ k,   // [B][C][N] bf16
           float* __restrict__ S) {                // [B][C][C] f32
    const int tid  = threadIdx.x;
    const int lane = tid & 31;
    const int wave = tid >> 5;
    const int msub = wave >> 2, nsub = wave & 3;
    int bi = blockIdx.x;
    const int nb = bi % (C_ / 256); bi /= (C_ / 256);
    const int cb = bi % (C_ / 32);  bi /= (C_ / 32);
    const int b  = bi;
    const int Mbase = cb * 32 + msub * 16;
    const int Nbase = nb * 256 + nsub * 64;
    const int l15 = lane & 15;
    const int khA = (lane >> 4) * 8;
    const int khB = (lane >> 4) * 16;
    const size_t cb64 = (size_t)b * C_ * N_;

    v8f acc[4];
#pragma unroll
    for (int i = 0; i < 4; ++i) acc[i] = vzero();

    const unsigned short* arow = q + cb64 + (size_t)(Mbase + l15) * N_;
    for (int kk = 0; kk < N_; kk += 32) {
        v16bf a = ldA(arow + kk, khA);
#pragma unroll
        for (int i = 0; i < 4; ++i) {
            const unsigned short* brow =
                k + cb64 + (size_t)(Nbase + i * 16 + l15) * N_ + kk + khB;
            acc[i] = WMMA_BF16(a, ldB(brow), acc[i]);
        }
    }

    const float scale = 1.0f / (float)C_;
    const int rowoff = (lane >> 4) * 8;
#pragma unroll
    for (int i = 0; i < 4; ++i) {
        const int col = Nbase + i * 16 + l15;
#pragma unroll
        for (int j = 0; j < 8; ++j) {
            const int row = Mbase + rowoff + j;
            S[(size_t)b * C_ * C_ + (size_t)row * C_ + col] = acc[i][j] * scale;
        }
    }
}

// ---------------------------------------------------------------------------
// K5: row softmax over S (rows of length C), write bf16 P.
// ---------------------------------------------------------------------------
__global__ void __launch_bounds__(256)
softmax_rows(const float* __restrict__ S, unsigned short* __restrict__ P) {
    __shared__ float red[256];
    const int row = blockIdx.x;                // b*C + c
    const int tid = threadIdx.x;
    const float* src = S + (size_t)row * C_;
    float x0 = src[tid], x1 = src[tid + 256];
    red[tid] = fmaxf(x0, x1);
    __syncthreads();
    for (int s = 128; s > 0; s >>= 1) {
        if (tid < s) red[tid] = fmaxf(red[tid], red[tid + s]);
        __syncthreads();
    }
    const float m = red[0];
    __syncthreads();
    float e0 = __expf(x0 - m), e1 = __expf(x1 - m);
    red[tid] = e0 + e1;
    __syncthreads();
    for (int s = 128; s > 0; s >>= 1) {
        if (tid < s) red[tid] += red[tid + s];
        __syncthreads();
    }
    const float inv = 1.0f / red[0];
    unsigned short* dst = P + (size_t)row * C_;
    dst[tid]       = f2bf(e0 * inv);
    dst[tid + 256] = f2bf(e1 * inv);
}

// ---------------------------------------------------------------------------
// K6: O[b][n][c] = sum_d P[b][c][d] * v[b][n][d]     (K = C = 512)
// ---------------------------------------------------------------------------
__global__ void __launch_bounds__(256)
attnv_gemm(const unsigned short* __restrict__ P,   // [B][C][C] bf16
           const unsigned short* __restrict__ v,   // [B][N][C] bf16
           unsigned short* __restrict__ O) {       // [B][N][C] bf16
    const int tid  = threadIdx.x;
    const int lane = tid & 31;
    const int wave = tid >> 5;
    const int msub = wave >> 2, nsub = wave & 3;
    int bi = blockIdx.x;
    const int nb = bi % (N_ / 256); bi /= (N_ / 256);
    const int cb = bi % (C_ / 32);  bi /= (C_ / 32);
    const int b  = bi;
    const int Mbase = cb * 32 + msub * 16;
    const int Nbase = nb * 256 + nsub * 64;
    const int l15 = lane & 15;
    const int khA = (lane >> 4) * 8;
    const int khB = (lane >> 4) * 16;
    const size_t pb = (size_t)b * C_ * C_;
    const size_t vb = (size_t)b * N_ * C_;

    v8f acc[4];
#pragma unroll
    for (int i = 0; i < 4; ++i) acc[i] = vzero();

    const unsigned short* arow = P + pb + (size_t)(Mbase + l15) * C_;
    for (int kk = 0; kk < C_; kk += 32) {
        v16bf a = ldA(arow + kk, khA);
#pragma unroll
        for (int i = 0; i < 4; ++i) {
            const unsigned short* brow =
                v + vb + (size_t)(Nbase + i * 16 + l15) * C_ + kk + khB;
            acc[i] = WMMA_BF16(a, ldB(brow), acc[i]);
        }
    }

    const int rowoff = (lane >> 4) * 8;
#pragma unroll
    for (int i = 0; i < 4; ++i) {
        const int n = Nbase + i * 16 + l15;
#pragma unroll
        for (int j = 0; j < 8; ++j) {
            const int row = Mbase + rowoff + j;
            O[vb + (size_t)n * C_ + row] = f2bf(acc[i][j]);
        }
    }
}

// ---------------------------------------------------------------------------
// K7: y[b][n][o] = sum_c wp[o][c] * O[b][n][c] + bias[o]   (f32 output)
// ---------------------------------------------------------------------------
__global__ void __launch_bounds__(256)
proj_gemm(const unsigned short* __restrict__ wp,   // [C][C] bf16
          const unsigned short* __restrict__ O,    // [B][N][C] bf16
          const float* __restrict__ bias,          // [C]
          float* __restrict__ y) {                 // [B][N][C] f32
    const int tid  = threadIdx.x;
    const int lane = tid & 31;
    const int wave = tid >> 5;
    const int msub = wave >> 2, nsub = wave & 3;
    int bi = blockIdx.x;
    const int nb = bi % (N_ / 256); bi /= (N_ / 256);
    const int cb = bi % (C_ / 32);  bi /= (C_ / 32);
    const int b  = bi;
    const int Mbase = cb * 32 + msub * 16;
    const int Nbase = nb * 256 + nsub * 64;
    const int l15 = lane & 15;
    const int khA = (lane >> 4) * 8;
    const int khB = (lane >> 4) * 16;
    const size_t ob = (size_t)b * N_ * C_;

    v8f acc[4];
#pragma unroll
    for (int i = 0; i < 4; ++i) acc[i] = vzero();

    const unsigned short* arow = wp + (size_t)(Mbase + l15) * C_;
    for (int kk = 0; kk < C_; kk += 32) {
        v16bf a = ldA(arow + kk, khA);
#pragma unroll
        for (int i = 0; i < 4; ++i) {
            const unsigned short* brow =
                O + ob + (size_t)(Nbase + i * 16 + l15) * C_ + kk + khB;
            acc[i] = WMMA_BF16(a, ldB(brow), acc[i]);
        }
    }

    const int rowoff = (lane >> 4) * 8;
    float bj[8];
#pragma unroll
    for (int j = 0; j < 8; ++j) bj[j] = bias[Mbase + rowoff + j];

#pragma unroll
    for (int i = 0; i < 4; ++i) {
        const int n = Nbase + i * 16 + l15;
#pragma unroll
        for (int j = 0; j < 8; ++j) {
            const int row = Mbase + rowoff + j;
            y[ob + (size_t)n * C_ + row] = acc[i][j] + bj[j];
        }
    }
}

// ---------------------------------------------------------------------------
extern "C" void kernel_launch(void* const* d_in, const int* in_sizes, int n_in,
                              void* d_out, int out_size, void* d_ws, size_t ws_size,
                              hipStream_t stream) {
    const float* x    = (const float*)d_in[0];
    const float* w1q  = (const float*)d_in[1];
    const float* w3q  = (const float*)d_in[2];
    const float* w1k  = (const float*)d_in[3];
    const float* w3k  = (const float*)d_in[4];
    const float* w1v  = (const float*)d_in[5];
    const float* w3v  = (const float*)d_in[6];
    const float* wprj = (const float*)d_in[7];
    const float* bprj = (const float*)d_in[8];
    float* y = (float*)d_out;

    // Workspace carve-up (bytes, all 256-aligned). Total ~189 MiB.
    char* ws = (char*)d_ws;
    size_t off = 0;
    unsigned short* xbf = (unsigned short*)(ws + off); off += (size_t)B_ * N_ * C_ * 2;  // 32 MiB
    unsigned short* We  = (unsigned short*)(ws + off); off += (size_t)9 * C_ * C_ * 2;   // 4.5 MiB
    unsigned short* wpb = (unsigned short*)(ws + off); off += (size_t)C_ * C_ * 2;       // 0.5 MiB
    unsigned short* q   = (unsigned short*)(ws + off); off += (size_t)B_ * C_ * N_ * 2;  // 32 MiB
    unsigned short* k   = (unsigned short*)(ws + off); off += (size_t)B_ * C_ * N_ * 2;  // 32 MiB
    unsigned short* v   = (unsigned short*)(ws + off); off += (size_t)B_ * N_ * C_ * 2;  // 32 MiB
    float*          S   = (float*)         (ws + off); off += (size_t)B_ * C_ * C_ * 4;  // 16 MiB
    unsigned short* P   = (unsigned short*)(ws + off); off += (size_t)B_ * C_ * C_ * 2;  // 8 MiB
    unsigned short* O   = (unsigned short*)(ws + off); off += (size_t)B_ * N_ * C_ * 2;  // 32 MiB
    (void)ws_size; (void)in_sizes; (void)n_in; (void)out_size;

    // K1: fold conv3 ∘ conv1 -> We[branch][t] (bf16)
    prep_we<<<(9 * C_ * C_) / 256, 256, 0, stream>>>(w1q, w3q, w1k, w3k, w1v, w3v, We);
    // K2: bf16 copies of x (kept in [B][N][C]!) and w_proj
    cvt_bf4<<<(B_ * N_ * C_ / 4 + 255) / 256, 256, 0, stream>>>(x, xbf, B_ * N_ * C_ / 4);
    cvt_bf4<<<(C_ * C_ / 4 + 255) / 256, 256, 0, stream>>>(wprj, wpb, C_ * C_ / 4);

    // K3: q, k ([B][C][N]) and v ([B][N][C]) via shifted-GEMM WMMA
    dim3 g3(B_ * (C_ / 32) * (N_ / 256));  // 2048 WGs
    qkv_gemm<<<g3, 256, 0, stream>>>(xbf, We + (size_t)0 * 3 * C_ * C_, q, 0);
    qkv_gemm<<<g3, 256, 0, stream>>>(xbf, We + (size_t)1 * 3 * C_ * C_, k, 0);
    qkv_gemm<<<g3, 256, 0, stream>>>(xbf, We + (size_t)2 * 3 * C_ * C_, v, 1);

    // K4: channel-attention scores (scale folded in)
    score_gemm<<<B_ * (C_ / 32) * (C_ / 256), 256, 0, stream>>>(q, k, S);
    // K5: softmax over last dim, bf16 probabilities
    softmax_rows<<<B_ * C_, 256, 0, stream>>>(S, P);
    // K6: attn @ v, stored [B][N][C]
    attnv_gemm<<<g3, 256, 0, stream>>>(P, v, O);
    // K7: projection + bias, f32 output in [B][N][C]
    proj_gemm<<<g3, 256, 0, stream>>>(wpb, O, bprj, y);
}
